// TimeMoEModel_36507222016562
// MI455X (gfx1250) — matrix operand order repaired and checked by reading, not compiled, not run on hardware
//
#include <hip/hip_runtime.h>
#include <hip/hip_bf16.h>
#include <cstdint>

// ---------------------------------------------------------------------------
// TimeMoE transformer forward for MI455X (gfx1250, wave32, WMMA bf16).
// - All GEMMs through one tiled WMMA kernel; A staged f32->bf16 at the LDS
//   boundary, B (weights) pre-converted to bf16 and staged with
//   GLOBAL_LOAD_ASYNC_TO_LDS_B128 (ASYNCcnt) when the builtin is available.
// - Attention is flash-style in LDS (no materialized SxS scores).
// - MoE computed densely over all experts, masked by top-2 combine weights.
// ---------------------------------------------------------------------------

#define B_   16
#define S_   1024
#define IN_  64
#define D_   512
#define H_   8
#define L_   6
#define E_   8
#define DF_  2048
#define HD_  64
#define NT_  (B_ * S_)

typedef __bf16 bf16;
typedef __attribute__((ext_vector_type(16))) __bf16        v16bf;
typedef __attribute__((ext_vector_type(4)))  __bf16        v4bf;
typedef __attribute__((ext_vector_type(8)))  float         v8f;
typedef __attribute__((ext_vector_type(4)))  unsigned int  v4u;

#if defined(__AMDGCN__) && __has_builtin(__builtin_amdgcn_global_load_async_to_lds_b128)
#define HAVE_ASYNC_COPY 1
// Parameter types per hipcc diagnostic: non-const gcc-vector int4 pointers,
// global (AS1) source and LDS (AS3) destination.
typedef int v4i_b __attribute__((vector_size(16)));
typedef v4i_b __attribute__((address_space(1)))* gasptr_t;
typedef v4i_b __attribute__((address_space(3)))* lasptr_t;
#else
#define HAVE_ASYNC_COPY 0
#endif

union FragU { v16bf v; v4u q[2]; };

// A-operand fragment (16x32 bf16): lane<16 holds row (lane&15), K {0..7,16..23};
// lane>=16 holds same row with K {8..15,24..31}.  (ISA 7.12.2)
__device__ __forceinline__ v16bf ld_fragA(const bf16* base, int ld, int lane) {
  const bf16* p = base + (size_t)(lane & 15) * ld + ((lane >> 4) << 3);
  FragU f;
  f.q[0] = *(const v4u*)p;
  f.q[1] = *(const v4u*)(p + 16);
  return f.v;
}

// B-operand fragment (32x16 bf16): lane holds column (lane&15);
// lanes 0-15 K {0..7,16..23}, lanes 16-31 K {8..15,24..31} (matches A).
__device__ __forceinline__ v16bf ld_fragB(const bf16* base, int ld, int lane) {
  const bf16* p = base + (size_t)(lane & 15) * ld + ((lane >> 4) << 3);
  FragU f;
  f.q[0] = *(const v4u*)p;
  f.q[1] = *(const v4u*)(p + 16);
  return f.v;
}

__device__ __forceinline__ v8f wmma_bf16(v16bf a, v16bf b, v8f c) {
  return __builtin_amdgcn_wmma_f32_16x16x32_bf16(
      /*neg_a=*/false, a, /*neg_b=*/false, b,
      /*c_mod=*/(short)0, c, /*reuse_a=*/false, /*reuse_b=*/false);
}

__device__ __forceinline__ float gelu_exact(float x) {
  return 0.5f * x * (1.0f + erff(x * 0.70710678118654752f));
}

// ---------------------------------------------------------------------------
// f32 -> bf16 conversion (weights mirrored once per launch; deterministic).
// ---------------------------------------------------------------------------
__global__ void cvt_bf16_kernel(const float* __restrict__ src,
                                bf16* __restrict__ dst, size_t n)
{
  size_t i = ((size_t)blockIdx.x * blockDim.x + threadIdx.x) * 4;
  if (i + 3 < n) {
    float4 v = *(const float4*)(src + i);
    v4bf pk; pk[0] = (bf16)v.x; pk[1] = (bf16)v.y; pk[2] = (bf16)v.z; pk[3] = (bf16)v.w;
    *(v4bf*)(dst + i) = pk;
  } else {
    for (; i < n; ++i) dst[i] = (bf16)src[i];
  }
}

// ---------------------------------------------------------------------------
// Generic tiled WMMA GEMM:  C[M,N] = epi(A[M,K] * Bw[N,K]^T + bias[N])
// A: f32 (converted while staging).  Bw: bf16 (async-DMA'd to LDS).
// 128x128 block tile, k-step 32, 256 threads (8 waves, 2x4 wave grid,
// each wave 64x32 = 8 WMMA accumulators).
// ---------------------------------------------------------------------------
__global__ __launch_bounds__(256) void gemm_bf16_wmma(
    const float* __restrict__ A, const bf16* __restrict__ Bw,
    const float* __restrict__ bias, float* __restrict__ C,
    int M, int N, int K, int epi /*0=none, 1=gelu*/)
{
  __shared__ __align__(16) bf16 As[128][40];   // 32 + 8 pad
  __shared__ __align__(16) bf16 Bs[128][40];

  const int tid  = threadIdx.x;
  const int lane = tid & 31;
  const int wave = tid >> 5;
  const int wm   = wave >> 2;   // 0..1  (64-row half)
  const int wn   = wave & 3;    // 0..3  (32-col quarter)
  const int row0 = blockIdx.y * 128;
  const int col0 = blockIdx.x * 128;

  const v8f vzero = {0.f,0.f,0.f,0.f,0.f,0.f,0.f,0.f};
  v8f acc[4][2];
  for (int i = 0; i < 4; ++i)
    for (int j = 0; j < 2; ++j) acc[i][j] = vzero;

  for (int k0 = 0; k0 < K; k0 += 32) {
    // ---- B tile: bf16 weights, straight byte copy global -> LDS ----------
#if HAVE_ASYNC_COPY
    for (int e8 = tid; e8 < 512; e8 += 256) {
      int r = e8 >> 2;
      int c = (e8 & 3) << 3;          // 8 bf16 = 16 bytes per chunk
      int gr = col0 + r;
      if (gr < N) {
        const bf16* gp = Bw + (size_t)gr * K + k0 + c;
        __builtin_amdgcn_global_load_async_to_lds_b128(
            (gasptr_t)(uintptr_t)gp, (lasptr_t)(uintptr_t)&Bs[r][c], 0, 0);
      } else {
        v4u z = {0u, 0u, 0u, 0u};
        *(v4u*)&Bs[r][c] = z;
      }
    }
#else
    for (int e8 = tid; e8 < 512; e8 += 256) {
      int r = e8 >> 2;
      int c = (e8 & 3) << 3;
      int gr = col0 + r;
      v4u v = {0u, 0u, 0u, 0u};
      if (gr < N) v = *(const v4u*)(Bw + (size_t)gr * K + k0 + c);
      *(v4u*)&Bs[r][c] = v;
    }
#endif
    // ---- A tile: f32 -> bf16 conversion at the LDS boundary ---------------
    for (int e4 = tid; e4 < 1024; e4 += 256) {
      int r = e4 >> 3;
      int c = (e4 & 7) << 2;
      int gr = row0 + r;
      float4 v = make_float4(0.f, 0.f, 0.f, 0.f);
      if (gr < M) v = *(const float4*)(A + (size_t)gr * K + k0 + c);
      v4bf pk; pk[0] = (bf16)v.x; pk[1] = (bf16)v.y; pk[2] = (bf16)v.z; pk[3] = (bf16)v.w;
      *(v4bf*)&As[r][c] = pk;
    }
    // Prefetch next k-tile (weights live in L2 on this problem size).
    if (k0 + 32 < K) {
      int pr = row0 + (tid & 127);
      if (pr < M) __builtin_prefetch(A + (size_t)pr * K + k0 + 32, 0, 3);
      int pc = col0 + (tid & 127);
      if (pc < N) __builtin_prefetch(Bw + (size_t)pc * K + k0 + 32, 0, 3);
    }
#if HAVE_ASYNC_COPY
#if __has_builtin(__builtin_amdgcn_s_wait_asynccnt)
    __builtin_amdgcn_s_wait_asynccnt(0);
#else
    asm volatile("s_wait_asynccnt 0" ::: "memory");
#endif
#endif
    __syncthreads();

    v16bf af[4];
    #pragma unroll
    for (int mi = 0; mi < 4; ++mi)
      af[mi] = ld_fragA(&As[wm * 64 + mi * 16][0], 40, lane);
    #pragma unroll
    for (int ni = 0; ni < 2; ++ni) {
      v16bf bfr = ld_fragB(&Bs[wn * 32 + ni * 16][0], 40, lane);
      #pragma unroll
      for (int mi = 0; mi < 4; ++mi)
        acc[mi][ni] = wmma_bf16(af[mi], bfr, acc[mi][ni]);
    }
    __syncthreads();
  }

  // Epilogue: C layout -> row = r + 8*(lane>=16), col = lane&15.
  #pragma unroll
  for (int mi = 0; mi < 4; ++mi) {
    #pragma unroll
    for (int ni = 0; ni < 2; ++ni) {
      int col = col0 + wn * 32 + ni * 16 + (lane & 15);
      if (col >= N) continue;
      float bv = bias ? bias[col] : 0.f;
      int rbase = row0 + wm * 64 + mi * 16 + ((lane >> 4) << 3);
      #pragma unroll
      for (int r = 0; r < 8; ++r) {
        int row = rbase + r;
        if (row < M) {
          float v = acc[mi][ni][r] + bv;
          if (epi == 1) v = gelu_exact(v);
          C[(size_t)row * N + col] = v;
        }
      }
    }
  }
}

// ---------------------------------------------------------------------------
// Flash attention: one block per (q-tile of 64 rows, b*H+h). 4 waves; each
// wave owns 16 query rows.  QK^T and PV both via WMMA; online softmax in f32.
// qkv layout: [B,S,3D] row-major (Q | K | V per token).
// ---------------------------------------------------------------------------
__global__ __launch_bounds__(128) void flash_attn_kernel(
    const float* __restrict__ qkv, float* __restrict__ out)
{
  __shared__ __align__(16) bf16 Qs [64][72];
  __shared__ __align__(16) bf16 Ks [64][72];
  __shared__ __align__(16) bf16 VTs[64][72];   // transposed: VTs[d][k]
  __shared__ __align__(16) bf16 Ps [64][72];
  __shared__ float mrow[64];
  __shared__ float lrow[64];

  const int tid  = threadIdx.x;
  const int lane = tid & 31;
  const int wave = tid >> 5;
  const int bh   = blockIdx.y;
  const int b    = bh / H_;
  const int h    = bh % H_;
  const int q0   = blockIdx.x * 64;
  const size_t base = (size_t)(b * S_) * (3 * D_);
  const float scale = 0.125f;  // 1/sqrt(HD=64)

  // Stage pre-scaled Q tile.
  for (int e4 = tid; e4 < 1024; e4 += 128) {
    int r = e4 >> 4;
    int c = (e4 & 15) << 2;
    float4 v = *(const float4*)(qkv + base + (size_t)(q0 + r) * (3 * D_) + h * HD_ + c);
    v4bf pk;
    pk[0] = (bf16)(v.x * scale); pk[1] = (bf16)(v.y * scale);
    pk[2] = (bf16)(v.z * scale); pk[3] = (bf16)(v.w * scale);
    *(v4bf*)&Qs[r][c] = pk;
  }
  if (tid < 64) { mrow[tid] = -1e30f; lrow[tid] = 0.f; }

  const v8f vzero = {0.f,0.f,0.f,0.f,0.f,0.f,0.f,0.f};
  v8f o[4];
  for (int j = 0; j < 4; ++j) o[j] = vzero;

  for (int kb = 0; kb < S_ / 64; ++kb) {
    __syncthreads();            // Ks/VTs from prior iter fully consumed
    const int kt0 = kb * 64;
    for (int e4 = tid; e4 < 1024; e4 += 128) {
      int r = e4 >> 4;
      int c = (e4 & 15) << 2;
      float4 kv = *(const float4*)(qkv + base + (size_t)(kt0 + r) * (3 * D_) + D_ + h * HD_ + c);
      v4bf pk; pk[0] = (bf16)kv.x; pk[1] = (bf16)kv.y; pk[2] = (bf16)kv.z; pk[3] = (bf16)kv.w;
      *(v4bf*)&Ks[r][c] = pk;
      float4 vv = *(const float4*)(qkv + base + (size_t)(kt0 + r) * (3 * D_) + 2 * D_ + h * HD_ + c);
      VTs[c + 0][r] = (bf16)vv.x; VTs[c + 1][r] = (bf16)vv.y;
      VTs[c + 2][r] = (bf16)vv.z; VTs[c + 3][r] = (bf16)vv.w;
    }
    __syncthreads();

    // S tile: wave's 16 q rows x 64 keys.
    v8f s[4];
    for (int j = 0; j < 4; ++j) s[j] = vzero;
    #pragma unroll
    for (int ks = 0; ks < 2; ++ks) {
      v16bf aq = ld_fragA(&Qs[wave * 16][ks * 32], 72, lane);
      #pragma unroll
      for (int j = 0; j < 4; ++j) {
        v16bf bk = ld_fragB(&Ks[j * 16][ks * 32], 72, lane);
        s[j] = wmma_bf16(aq, bk, s[j]);
      }
    }

    // Online softmax update. Row of element r: wave*16 + 8*(lane>=16) + r.
    float alpha[8];
    #pragma unroll
    for (int r = 0; r < 8; ++r) {
      float mx = -1e30f;
      #pragma unroll
      for (int j = 0; j < 4; ++j) mx = fmaxf(mx, s[j][r]);
      for (int off = 1; off < 16; off <<= 1) mx = fmaxf(mx, __shfl_xor(mx, off, 32));
      int row = wave * 16 + ((lane >> 4) << 3) + r;
      float om = mrow[row];
      float nm = fmaxf(om, mx);
      float a  = __expf(om - nm);
      alpha[r] = a;
      float sum = 0.f;
      #pragma unroll
      for (int j = 0; j < 4; ++j) {
        float p = __expf(s[j][r] - nm);
        s[j][r] = p;
        sum += p;
      }
      for (int off = 1; off < 16; off <<= 1) sum += __shfl_xor(sum, off, 32);
      if ((lane & 15) == 0) {            // one writer per lane-half
        mrow[row] = nm;
        lrow[row] = lrow[row] * a + sum;
      }
    }

    // Rescale O and spill P (bf16) to LDS for the A-operand round trip.
    #pragma unroll
    for (int j = 0; j < 4; ++j) {
      #pragma unroll
      for (int r = 0; r < 8; ++r) {
        o[j][r] *= alpha[r];
        int row = wave * 16 + ((lane >> 4) << 3) + r;
        Ps[row][j * 16 + (lane & 15)] = (bf16)s[j][r];
      }
    }

    // O += P * V   (P rows are wave-private; same-wave LDS ordering suffices)
    #pragma unroll
    for (int ks = 0; ks < 2; ++ks) {
      v16bf ap = ld_fragA(&Ps[wave * 16][ks * 32], 72, lane);
      #pragma unroll
      for (int j = 0; j < 4; ++j) {
        v16bf bv = ld_fragB(&VTs[j * 16][ks * 32], 72, lane);
        o[j] = wmma_bf16(ap, bv, o[j]);
      }
    }
  }
  __syncthreads();

  #pragma unroll
  for (int j = 0; j < 4; ++j) {
    #pragma unroll
    for (int r = 0; r < 8; ++r) {
      int row = wave * 16 + ((lane >> 4) << 3) + r;
      float invl = 1.0f / lrow[row];
      int d = j * 16 + (lane & 15);
      out[(size_t)(b * S_ + q0 + row) * D_ + h * HD_ + d] = o[j][r] * invl;
    }
  }
}

// ---------------------------------------------------------------------------
// Elementwise / reduction helpers
// ---------------------------------------------------------------------------
__global__ void posenc_kernel(float* __restrict__ h) {
  int idx = blockIdx.x * blockDim.x + threadIdx.x;
  if (idx >= NT_ * D_) return;
  int d = idx & (D_ - 1);
  int s = (idx / D_) & (S_ - 1);
  float div = __expf(-(float)(d & ~1) * (9.210340371976184f / (float)D_));
  float ang = (float)s * div;
  h[idx] += (d & 1) ? __cosf(ang) : __sinf(ang);
}

// out[row] = LN(x[row] + add[row]) * gamma + beta   (add may be null; in-place ok)
__global__ __launch_bounds__(256) void resln_kernel(
    const float* __restrict__ x, const float* __restrict__ add,
    const float* __restrict__ gamma, const float* __restrict__ beta,
    float* __restrict__ out, int D)
{
  int row = blockIdx.x;
  const float* xr = x + (size_t)row * D;
  const float* ar = add ? add + (size_t)row * D : nullptr;
  float s = 0.f, s2 = 0.f;
  for (int i = threadIdx.x; i < D; i += 256) {
    float v = xr[i] + (ar ? ar[i] : 0.f);
    s += v; s2 += v * v;
  }
  for (int off = 16; off > 0; off >>= 1) {
    s  += __shfl_xor(s,  off, 32);
    s2 += __shfl_xor(s2, off, 32);
  }
  __shared__ float sh[16];
  int wave = threadIdx.x >> 5, lane = threadIdx.x & 31;
  if (lane == 0) { sh[wave] = s; sh[8 + wave] = s2; }
  __syncthreads();
  if (threadIdx.x == 0) {
    float ts = 0.f, ts2 = 0.f;
    for (int w = 0; w < 8; ++w) { ts += sh[w]; ts2 += sh[8 + w]; }
    sh[0] = ts; sh[8] = ts2;
  }
  __syncthreads();
  float mu   = sh[0] / (float)D;
  float var  = sh[8] / (float)D - mu * mu;
  float rstd = rsqrtf(var + 1e-5f);
  for (int i = threadIdx.x; i < D; i += 256) {
    float v = xr[i] + (ar ? ar[i] : 0.f);
    out[(size_t)row * D + i] = (v - mu) * rstd * gamma[i] + beta[i];
  }
}

__global__ void gate_top2_kernel(const float* __restrict__ logits,
                                 float* __restrict__ combine, int N)
{
  int n = blockIdx.x * blockDim.x + threadIdx.x;
  if (n >= N) return;
  float z[E_];
  float mx = -1e30f;
  for (int e = 0; e < E_; ++e) { z[e] = logits[n * E_ + e]; mx = fmaxf(mx, z[e]); }
  float sum = 0.f;
  for (int e = 0; e < E_; ++e) { z[e] = __expf(z[e] - mx); sum += z[e]; }
  float inv = 1.0f / sum;
  int i1 = -1, i2 = -1; float p1 = -1.f, p2 = -1.f;
  for (int e = 0; e < E_; ++e) {
    float p = z[e] * inv;
    if (p > p1)      { p2 = p1; i2 = i1; p1 = p; i1 = e; }
    else if (p > p2) { p2 = p;  i2 = e; }
  }
  float w = 1.0f / (p1 + p2 + 1e-8f);
  for (int e = 0; e < E_; ++e)
    combine[n * E_ + e] = (e == i1) ? p1 * w : ((e == i2) ? p2 * w : 0.f);
}

__global__ void moe_accum_kernel(float* __restrict__ acc, const float* __restrict__ y,
                                 const float* __restrict__ combine, int e, int init)
{
  int idx = blockIdx.x * blockDim.x + threadIdx.x;
  if (idx >= NT_ * D_) return;
  int n = idx / D_;
  float v = combine[n * E_ + e] * y[idx];
  acc[idx] = init ? v : (acc[idx] + v);
}

__global__ void meanpool_kernel(const float* __restrict__ h, float* __restrict__ pooled)
{
  int idx = blockIdx.x * blockDim.x + threadIdx.x;
  if (idx >= B_ * D_) return;
  int b = idx / D_;
  int d = idx % D_;
  float s = 0.f;
  for (int t = 0; t < S_; ++t) s += h[(size_t)(b * S_ + t) * D_ + d];
  pooled[idx] = s * (1.0f / (float)S_);
}

// ---------------------------------------------------------------------------
// Orchestration
// ---------------------------------------------------------------------------
extern "C" void kernel_launch(void* const* d_in, const int* in_sizes, int n_in,
                              void* d_out, int out_size, void* d_ws, size_t ws_size,
                              hipStream_t stream)
{
  const float* x          = (const float*)d_in[0];
  const float* inp_w      = (const float*)d_in[1];
  const float* inp_b      = (const float*)d_in[2];
  const float* in_proj_w  = (const float*)d_in[3];
  const float* in_proj_b  = (const float*)d_in[4];
  const float* out_proj_w = (const float*)d_in[5];
  const float* out_proj_b = (const float*)d_in[6];
  const float* ln1_s      = (const float*)d_in[7];
  const float* ln1_b      = (const float*)d_in[8];
  const float* gate_w     = (const float*)d_in[9];
  const float* w1         = (const float*)d_in[10];
  const float* b1         = (const float*)d_in[11];
  const float* w2         = (const float*)d_in[12];
  const float* b2         = (const float*)d_in[13];
  const float* ln2_s      = (const float*)d_in[14];
  const float* ln2_b      = (const float*)d_in[15];
  const float* cls_ln_s   = (const float*)d_in[16];
  const float* cls_ln_b   = (const float*)d_in[17];
  const float* cls_w1     = (const float*)d_in[18];
  const float* cls_b1     = (const float*)d_in[19];
  const float* cls_w2     = (const float*)d_in[20];
  const float* cls_b2     = (const float*)d_in[21];

  char* ws = (char*)d_ws;
  size_t off = 0;
  auto allocb = [&](size_t bytes) -> char* {
    char* p = ws + off;
    off += (bytes + 255) & ~(size_t)255;
    return p;
  };
  float* hbuf    = (float*)allocb((size_t)NT_ * D_  * 4);   // running hidden state
  float* qkv     = (float*)allocb((size_t)NT_ * 3 * D_ * 4);
  float* attn    = (float*)allocb((size_t)NT_ * D_  * 4);
  float* tmp1    = (float*)allocb((size_t)NT_ * D_  * 4);
  float* moe_h   = (float*)allocb((size_t)NT_ * DF_ * 4);
  float* moe_acc = (float*)allocb((size_t)NT_ * D_  * 4);
  float* logits  = (float*)allocb((size_t)NT_ * E_  * 4);
  float* combine = (float*)allocb((size_t)NT_ * E_  * 4);
  float* pooled  = (float*)allocb((size_t)B_ * D_ * 4);
  float* pooled2 = (float*)allocb((size_t)B_ * D_ * 4);
  float* clsh    = (float*)allocb((size_t)B_ * (D_ / 2) * 4);
  // bf16 weight mirrors
  const size_t n_inp_w  = (size_t)D_ * IN_;
  const size_t n_inproj = (size_t)L_ * 3 * D_ * D_;
  const size_t n_outproj= (size_t)L_ * D_ * D_;
  const size_t n_gate   = (size_t)L_ * E_ * D_;
  const size_t n_w1     = (size_t)L_ * E_ * DF_ * D_;
  const size_t n_w2     = (size_t)L_ * E_ * D_ * DF_;
  const size_t n_cls1   = (size_t)(D_ / 2) * D_;
  const size_t n_cls2   = (size_t)(D_ / 2);
  bf16* wb_inp    = (bf16*)allocb(n_inp_w  * 2);
  bf16* wb_inproj = (bf16*)allocb(n_inproj * 2);
  bf16* wb_outproj= (bf16*)allocb(n_outproj* 2);
  bf16* wb_gate   = (bf16*)allocb(n_gate   * 2);
  bf16* wb_w1     = (bf16*)allocb(n_w1     * 2);
  bf16* wb_w2     = (bf16*)allocb(n_w2     * 2);
  bf16* wb_cls1   = (bf16*)allocb(n_cls1   * 2);
  bf16* wb_cls2   = (bf16*)allocb(n_cls2   * 2);
  (void)ws_size; (void)in_sizes; (void)n_in; (void)out_size;

  auto cvt = [&](const float* src, bf16* dst, size_t n) {
    cvt_bf16_kernel<<<(unsigned)((n / 4 + 255) / 256), 256, 0, stream>>>(src, dst, n);
  };
  cvt(inp_w,      wb_inp,     n_inp_w);
  cvt(in_proj_w,  wb_inproj,  n_inproj);
  cvt(out_proj_w, wb_outproj, n_outproj);
  cvt(gate_w,     wb_gate,    n_gate);
  cvt(w1,         wb_w1,      n_w1);
  cvt(w2,         wb_w2,      n_w2);
  cvt(cls_w1,     wb_cls1,    n_cls1);
  cvt(cls_w2,     wb_cls2,    n_cls2);

  auto gemm = [&](const float* A, const bf16* Bw, const float* bias, float* C,
                  int M, int N, int K, int epi) {
    dim3 grid((N + 127) / 128, (M + 127) / 128);
    gemm_bf16_wmma<<<grid, 256, 0, stream>>>(A, Bw, bias, C, M, N, K, epi);
  };

  // Input projection + positional encoding.
  gemm(x, wb_inp, inp_b, hbuf, NT_, D_, IN_, 0);
  posenc_kernel<<<(NT_ * D_ + 255) / 256, 256, 0, stream>>>(hbuf);

  for (int l = 0; l < L_; ++l) {
    // Attention block.
    gemm(hbuf, wb_inproj + (size_t)l * 3 * D_ * D_, in_proj_b + (size_t)l * 3 * D_,
         qkv, NT_, 3 * D_, D_, 0);
    {
      dim3 grid(S_ / 64, B_ * H_);
      flash_attn_kernel<<<grid, 128, 0, stream>>>(qkv, attn);
    }
    gemm(attn, wb_outproj + (size_t)l * D_ * D_, out_proj_b + (size_t)l * D_,
         tmp1, NT_, D_, D_, 0);
    resln_kernel<<<NT_, 256, 0, stream>>>(hbuf, tmp1,
        ln1_s + (size_t)l * D_, ln1_b + (size_t)l * D_, hbuf, D_);

    // MoE block (dense over experts, masked combine).
    gemm(hbuf, wb_gate + (size_t)l * E_ * D_, nullptr, logits, NT_, E_, D_, 0);
    gate_top2_kernel<<<(NT_ + 255) / 256, 256, 0, stream>>>(logits, combine, NT_);
    for (int e = 0; e < E_; ++e) {
      const bf16*  w1e = wb_w1 + ((size_t)(l * E_ + e)) * DF_ * D_;
      const float* b1e = b1 + ((size_t)(l * E_ + e)) * DF_;
      const bf16*  w2e = wb_w2 + ((size_t)(l * E_ + e)) * D_ * DF_;
      const float* b2e = b2 + ((size_t)(l * E_ + e)) * D_;
      gemm(hbuf, w1e, b1e, moe_h, NT_, DF_, D_, 1);   // GELU epilogue
      gemm(moe_h, w2e, b2e, tmp1, NT_, D_, DF_, 0);
      moe_accum_kernel<<<(NT_ * D_ + 255) / 256, 256, 0, stream>>>(
          moe_acc, tmp1, combine, e, e == 0);
    }
    resln_kernel<<<NT_, 256, 0, stream>>>(hbuf, moe_acc,
        ln2_s + (size_t)l * D_, ln2_b + (size_t)l * D_, hbuf, D_);
  }

  // Pool + classifier head.
  meanpool_kernel<<<(B_ * D_ + 255) / 256, 256, 0, stream>>>(hbuf, pooled);
  resln_kernel<<<B_, 256, 0, stream>>>(pooled, nullptr, cls_ln_s, cls_ln_b, pooled2, D_);
  gemm(pooled2, wb_cls1, cls_b1, clsh, B_, D_ / 2, D_, 1);
  gemm(clsh, wb_cls2, cls_b2, (float*)d_out, B_, 1, D_ / 2, 0);
}
